// Generator_39874476376547
// MI455X (gfx1250) — compile-verified, matrix operand order checked by reading, not run on hardware
//
#include <hip/hip_runtime.h>

// ---------------- problem constants ----------------
#define BB 8192   // batch rows (GEMM M)
#define NN 2048   // nodes (GEMM N and K)
#define KK 4      // neighbors

typedef __attribute__((ext_vector_type(16))) _Float16 v16h;
typedef __attribute__((ext_vector_type(8)))  _Float16 v8h;
typedef __attribute__((ext_vector_type(4)))  _Float16 v4h;
typedef __attribute__((ext_vector_type(8)))  float    v8f;
typedef __attribute__((ext_vector_type(4)))  float    v4f;
typedef __attribute__((ext_vector_type(4)))  int      v4i;

typedef __attribute__((address_space(1))) v4i* global_v4i_ptr;
typedef __attribute__((address_space(3))) v4i* lds_v4i_ptr;

// --------- gfx1250 async global->LDS (ASYNCcnt) feature detection ----------
#if defined(__has_builtin)
#if __has_builtin(__builtin_amdgcn_global_load_async_to_lds_b128)
#define HAVE_ASYNC_LDS 1
#endif
#if __has_builtin(__builtin_amdgcn_s_wait_asynccnt)
#define HAVE_WAIT_ASYNC_BUILTIN 1
#endif
#endif
#ifndef HAVE_ASYNC_LDS
#define HAVE_ASYNC_LDS 0
#endif
#ifndef HAVE_WAIT_ASYNC_BUILTIN
#define HAVE_WAIT_ASYNC_BUILTIN 0
#endif

// 16-byte global->LDS copy. Async path goes straight to LDS (no VGPR round
// trip, tracked on ASYNCcnt); fallback stages through registers (still correct
// in the ping-pong loop).
__device__ __forceinline__ void cp16(const _Float16* __restrict__ g,
                                     _Float16* __restrict__ l) {
#if HAVE_ASYNC_LDS
    // signature: (v4i addrspace(1)* src, v4i addrspace(3)* dst, imm off, imm cpol)
    // flat LDS address low 32 bits == LDS byte offset (ISA 10.2 aperture)
    __builtin_amdgcn_global_load_async_to_lds_b128(
        (global_v4i_ptr)(uintptr_t)g,
        (lds_v4i_ptr)(uint32_t)(uintptr_t)l,
        0, 0);
#else
    *(v8h*)l = *(const v8h*)g;
#endif
}

__device__ __forceinline__ void wait_async_copies() {
#if HAVE_ASYNC_LDS
#if HAVE_WAIT_ASYNC_BUILTIN
    __builtin_amdgcn_s_wait_asynccnt(0);
#else
    asm volatile("s_wait_asynccnt 0" ::: "memory");
#endif
#endif
}

// ---------------- kernel 1: lin_w f32 -> f16 ----------------
__global__ __launch_bounds__(256) void wconvert_kernel(const float* __restrict__ w,
                                                       _Float16* __restrict__ wh) {
    size_t i = ((size_t)blockIdx.x * blockDim.x + threadIdx.x) * 4;
    v4f v = *(const v4f*)(w + i);
    v4h h;
    h[0] = (_Float16)v[0]; h[1] = (_Float16)v[1];
    h[2] = (_Float16)v[2]; h[3] = (_Float16)v[3];
    *(v4h*)(wh + i) = h;
}

// ---------------- kernel 2: fused gather + 1x4 conv + relu -> Y (f16) ------
// Y[b2, p] = relu(cb + sum_j x[s*2048+p, adj[q,j]]*w0[j] + label[..]*w1[j])
// with q = b2/4, s = b2%4.
__global__ __launch_bounds__(256) void y_kernel(const float* __restrict__ x,
                                                const float* __restrict__ label,
                                                const int*   __restrict__ adj,
                                                const float* __restrict__ conv_w,
                                                const float* __restrict__ conv_b,
                                                _Float16* __restrict__ Y) {
    const int b2 = blockIdx.x;
    const int q  = b2 >> 2;
    const int s  = b2 & 3;

    const int a0 = adj[q * KK + 0];
    const int a1 = adj[q * KK + 1];
    const int a2 = adj[q * KK + 2];
    const int a3 = adj[q * KK + 3];

    const float w0 = conv_w[0], w1 = conv_w[1], w2 = conv_w[2], w3 = conv_w[3];
    const float u0 = conv_w[4], u1 = conv_w[5], u2 = conv_w[6], u3 = conv_w[7];
    const float cb = conv_b[0];

    const float* __restrict__ xb = x     + (size_t)s * NN * NN;
    const float* __restrict__ lb = label + (size_t)s * NN * NN;
    _Float16* __restrict__ yrow  = Y + (size_t)b2 * NN;

    for (int p = threadIdx.x; p < NN; p += blockDim.x) {
        const float* __restrict__ xr = xb + (size_t)p * NN;
        const float* __restrict__ lr = lb + (size_t)p * NN;
        float acc = cb;
        acc += xr[a0] * w0 + xr[a1] * w1 + xr[a2] * w2 + xr[a3] * w3;
        acc += lr[a0] * u0 + lr[a1] * u1 + lr[a2] * u2 + lr[a3] * u3;
        acc = fmaxf(acc, 0.0f);
        yrow[p] = (_Float16)acc;
    }
}

// ---------------- kernel 3: WMMA GEMM  out = Y(8192x2048) * Wh^T + bias ----
#define M_TILE 128    // 2 waves x 64
#define N_TILE 256    // 4 waves x 64
#define K_TILE 32
#define LDW    40     // padded LDS row stride in halfs (80B) for conflict-free b128 reads
#define A_BUF  (M_TILE * LDW)   // 5120 halfs = 10 KB
#define B_BUF  (N_TILE * LDW)   // 10240 halfs = 20 KB

// Fragment load per CDNA5 16-bit A/B VGPR layout (ISA 7.12.2):
// lanes 0-15:  halfs = K[0..7] , K[16..23]  of row (lane&15)
// lanes 16-31: halfs = K[8..15], K[24..31] of row (lane&15)
__device__ __forceinline__ v16h ld_frag(const _Float16* __restrict__ tile, int lane) {
    const int r  = lane & 15;
    const int ko = (lane >> 4) << 3;
    const _Float16* p = tile + r * LDW + ko;
    v8h lo = *(const v8h*)(p);
    v8h hi = *(const v8h*)(p + 16);
    v16h v;
#pragma unroll
    for (int i = 0; i < 8; ++i) { v[i] = lo[i]; v[i + 8] = hi[i]; }
    return v;
}

__global__ __launch_bounds__(256) void gemm_kernel(const _Float16* __restrict__ Y,
                                                   const _Float16* __restrict__ Wh,
                                                   const float* __restrict__ lin_b,
                                                   float* __restrict__ out) {
    __shared__ __align__(16) _Float16 ldsA[2 * A_BUF]; // 20 KB
    __shared__ __align__(16) _Float16 ldsB[2 * B_BUF]; // 40 KB

    const int tid  = threadIdx.x;
    const int lane = tid & 31;
    const int wv   = tid >> 5;      // wave 0..7
    const int wm   = wv & 1;        // 2 waves along M
    const int wn   = wv >> 1;       // 4 waves along N
    const int m0   = blockIdx.x * M_TILE;
    const int n0   = blockIdx.y * N_TILE;

    // stage one K_TILE slab (A: 128x32 halfs, B: 256x32 halfs) into buffer `buf`
    auto stage = [&](int buf, int kb) {
        _Float16* sA = ldsA + buf * A_BUF;
        _Float16* sB = ldsB + buf * B_BUF;
#pragma unroll
        for (int k = 0; k < 2; ++k) {           // 512 x 16B for A
            const int u = tid + 256 * k;
            const int r = u >> 2, f = (u & 3) * 8;
            cp16(Y + (size_t)(m0 + r) * NN + kb + f, sA + r * LDW + f);
        }
#pragma unroll
        for (int k = 0; k < 4; ++k) {           // 1024 x 16B for B
            const int u = tid + 256 * k;
            const int r = u >> 2, f = (u & 3) * 8;
            cp16(Wh + (size_t)(n0 + r) * NN + kb + f, sB + r * LDW + f);
        }
    };

    v8f acc[4][4] = {};

    // prologue: fill buffer 0
    stage(0, 0);
    wait_async_copies();
    __syncthreads();

    for (int step = 0; step < NN / K_TILE; ++step) {
        const int cur = step & 1;
        const int kb_next = (step + 1) * K_TILE;
        if (kb_next < NN) stage(cur ^ 1, kb_next);   // prefetch next slab

        const _Float16* sA = ldsA + cur * A_BUF + (wm * 64) * LDW;
        const _Float16* sB = ldsB + cur * B_BUF + (wn * 64) * LDW;

        v16h af[4], bf[4];
#pragma unroll
        for (int i = 0; i < 4; ++i) af[i] = ld_frag(sA + i * 16 * LDW, lane);
#pragma unroll
        for (int j = 0; j < 4; ++j) bf[j] = ld_frag(sB + j * 16 * LDW, lane);

#pragma unroll
        for (int i = 0; i < 4; ++i)
#pragma unroll
            for (int j = 0; j < 4; ++j)
                acc[i][j] = __builtin_amdgcn_wmma_f32_16x16x32_f16(
                    false, af[i], false, bf[j], (short)0, acc[i][j], false, false);

        wait_async_copies();   // our writes into buf^1 have landed in LDS
        __syncthreads();       // everyone done reading buf, writes visible
    }

    // epilogue: D element (M = r + (lane>=16)*8, N = lane&15) per ISA 7.12.2
    const int msub = (lane >> 4) << 3;
    const int nidx = lane & 15;

#pragma unroll
    for (int j = 0; j < 4; ++j) {
        const int n = n0 + wn * 64 + j * 16 + nidx;
        const float bias = lin_b[n];
#pragma unroll
        for (int i = 0; i < 4; ++i) {
            const int mb = m0 + wm * 64 + i * 16 + msub;
#pragma unroll
            for (int r = 0; r < 8; ++r)
                out[(size_t)(mb + r) * NN + n] = acc[i][j][r] + bias;
        }
    }
}

// ---------------- launcher ----------------
extern "C" void kernel_launch(void* const* d_in, const int* in_sizes, int n_in,
                              void* d_out, int out_size, void* d_ws, size_t ws_size,
                              hipStream_t stream) {
    const float* x      = (const float*)d_in[0];
    const float* label  = (const float*)d_in[1];
    const int*   adj    = (const int*)  d_in[2];
    const float* conv_w = (const float*)d_in[3];
    const float* conv_b = (const float*)d_in[4];
    const float* lin_w  = (const float*)d_in[5];
    const float* lin_b  = (const float*)d_in[6];
    float* out = (float*)d_out;

    // workspace: Y (f16, 32 MB) then Wh (f16, 8 MB)
    _Float16* Yh = (_Float16*)d_ws;
    _Float16* Wh = (_Float16*)((char*)d_ws + (size_t)BB * NN * sizeof(_Float16));

    // 1) lin_w -> f16  (4M elems, 4 per thread)
    wconvert_kernel<<<(NN * NN) / (256 * 4), 256, 0, stream>>>(lin_w, Wh);

    // 2) fused gather + conv + relu -> Y f16
    y_kernel<<<BB, 256, 0, stream>>>(x, label, adj, conv_w, conv_b, Yh);

    // 3) double-buffered WMMA GEMM with bias, async global->LDS staging
    dim3 grid(BB / M_TILE, NN / N_TILE); // (64, 8)
    gemm_kernel<<<grid, 256, 0, stream>>>(Yh, Wh, lin_b, out);
}
